// EchoStateNetwork_13039520710687
// MI455X (gfx1250) — matrix operand order repaired and checked by reading, not compiled, not run on hardware
//
#include <hip/hip_runtime.h>
#include <hip/hip_bf16.h>
#include <math.h>

// Problem constants (match reference)
#define B_ 32
#define T_ 1000
#define I_ 3
#define R_ 2048
#define O_ 3

#define NBLK 64      // persistent workgroups (co-resident; 4 wave32 each = 256 tiles)
#define NTHR 128

typedef __bf16 bf16_t;
typedef bf16_t v16bf __attribute__((ext_vector_type(16)));
typedef bf16_t v8bf  __attribute__((ext_vector_type(8)));
typedef float  v8f   __attribute__((ext_vector_type(8)));

// ---------------- workspace layout (bytes) ----------------
#define WS_WB   0u                      // bf16 Wres: R*R*2 = 8 MB (L2-resident)
#define WS_SB0  (8u*1024u*1024u)        // bf16 state ping: B*R*2 = 128 KB
#define WS_SB1  (WS_SB0 + 131072u)      // bf16 state pong
#define WS_SF   (WS_SB1 + 131072u)      // fp32 final state: B*R*4 = 256 KB
#define WS_BAR  (WS_SF  + 262144u)      // 2 x u32 barrier state

// ---------------- setup: Wres -> bf16, zero state, reset barrier ----------------
__global__ void esn_setup(const float* __restrict__ Wres, bf16_t* __restrict__ Wb,
                          bf16_t* __restrict__ Sb0, bf16_t* __restrict__ Sb1,
                          unsigned* __restrict__ bar) {
  int idx = blockIdx.x * blockDim.x + threadIdx.x;
  if (idx < R_ * R_) Wb[idx] = (bf16_t)Wres[idx];
  if (idx < B_ * R_) { Sb0[idx] = (bf16_t)0.0f; Sb1[idx] = (bf16_t)0.0f; }
  if (idx == 0)      { bar[0] = 0u; bar[1] = 0u; }
}

// ---------------- device-wide sense-reversing barrier ----------------
__device__ __forceinline__ void grid_barrier(unsigned* count, unsigned* gen, unsigned nblk) {
  __threadfence();        // release: every thread's stores visible at device scope
  __syncthreads();
  if (threadIdx.x == 0) {
    unsigned g = __hip_atomic_load(gen, __ATOMIC_RELAXED, __HIP_MEMORY_SCOPE_AGENT);
    unsigned arrived =
        __hip_atomic_fetch_add(count, 1u, __ATOMIC_ACQ_REL, __HIP_MEMORY_SCOPE_AGENT);
    if (arrived == nblk - 1u) {
      __hip_atomic_store(count, 0u, __ATOMIC_RELAXED, __HIP_MEMORY_SCOPE_AGENT);
      __hip_atomic_store(gen, g + 1u, __ATOMIC_RELEASE, __HIP_MEMORY_SCOPE_AGENT);
    } else {
      while (__hip_atomic_load(gen, __ATOMIC_ACQUIRE, __HIP_MEMORY_SCOPE_AGENT) == g) {
        __builtin_amdgcn_s_sleep(2);
      }
    }
  }
  __syncthreads();
  __threadfence();        // acquire side for all waves: invalidate near caches
}

// ---------------- persistent recurrence kernel ----------------
// Tile map: block owns a 32-column N slice; wave w -> (m_tile = w&1, n_tile = w>>1).
// D = tanh( S(bf16) x Wres^T(bf16) + noise + x_t Win^T ), f32 accum via WMMA.
__global__ void __launch_bounds__(NTHR, 1)
esn_steps(const float* __restrict__ x,      // [B,T,I]
          const float* __restrict__ Win,    // [R,I]
          const float* __restrict__ noise,  // [B,R]
          const bf16_t* __restrict__ Wb,    // [R,R] bf16
          bf16_t* __restrict__ Sb0,
          bf16_t* __restrict__ Sb1,
          float* __restrict__ Sf,           // [B,R] fp32 final state
          float* __restrict__ states,       // [B,T,R] fp32 output (streaming)
          unsigned* __restrict__ bar) {
  const int lane  = threadIdx.x & 31;
  const int wave  = threadIdx.x >> 5;
  const int m0    = (wave & 1) * 16;
  const int n0    = blockIdx.x * 32 + (wave >> 1) * 16;
  const int nloc  = lane & 15;
  const int n     = n0 + nloc;                       // lane's reservoir column (N)
  const int mhalf = (lane >> 4) * 8;                 // D layout: lanes 16-31 -> M+8

  // Hoisted per-lane constants (invariant across t).
  const float w0 = Win[n * I_ + 0];
  const float w1 = Win[n * I_ + 1];
  const float w2 = Win[n * I_ + 2];
  float bias[8];
#pragma unroll
  for (int v = 0; v < 8; ++v) bias[v] = noise[(m0 + mhalf + v) * R_ + n];

  // Fragment addressing: contiguous K order (k = 16*(lane/16) + s), same for A and B.
  const int aoff = (lane < 16) ? 0 : 16;   // A: K {0..15} vs {16..31}
  const int boff = (lane < 16) ? 0 : 16;   // B: K low half vs high half
  const int m_a  = m0 + (lane & 15);       // A: M = lane%16
  const bf16_t* __restrict__ Brow = Wb + (size_t)n * R_;

  for (int t = 0; t < T_; ++t) {
    const bf16_t* __restrict__ Sc = (t & 1) ? Sb1 : Sb0;  // read
    bf16_t* __restrict__       Sn = (t & 1) ? Sb0 : Sb1;  // write (ping-pong)
    const bf16_t* __restrict__ Arow = Sc + m_a * R_;

    // Accumulator init: noise + x_t @ Win^T (K=3 input projection, scalar).
    v8f c;
#pragma unroll
    for (int v = 0; v < 8; ++v) {
      const int m = m0 + mhalf + v;
      const float* xp = x + ((size_t)m * T_ + t) * I_;
      c[v] = bias[v] + xp[0] * w0 + xp[1] * w1 + xp[2] * w2;
    }

    // K loop: 64 x v_wmma_f32_16x16x32_bf16; A/B stream from L2.
    for (int k0 = 0; k0 < R_; k0 += 32) {
      v8bf alo = *(const v8bf*)(Arow + k0 + aoff);
      v8bf ahi = *(const v8bf*)(Arow + k0 + 8 + aoff);
      v16bf a;
#pragma unroll
      for (int j = 0; j < 8; ++j) { a[j] = alo[j]; a[8 + j] = ahi[j]; }
      v16bf b = *(const v16bf*)(Brow + k0 + boff);
      __builtin_prefetch(Brow + k0 + 32 + boff, 0, 3);   // global_prefetch_b8
      c = __builtin_amdgcn_wmma_f32_16x16x32_bf16(
              /*neg_a=*/false, a, /*neg_b=*/false, b,
              /*c_mod=*/(short)0, c, /*reuse_a=*/false, /*reuse_b=*/false);
    }

    // Epilogue: tanh; NT-store the 256 MB streaming states (don't pollute L2,
    // keep Wb resident); bf16 next-state stays regular-temporal (L2-resident).
#pragma unroll
    for (int v = 0; v < 8; ++v) {
      const int m = m0 + mhalf + v;
      const float val = tanhf(c[v]);
      __builtin_nontemporal_store(val, &states[((size_t)m * T_ + t) * R_ + n]);
      Sn[m * R_ + n] = (bf16_t)val;
      if (t == T_ - 1) Sf[m * R_ + n] = val;
    }

    grid_barrier(bar, bar + 1, (unsigned)gridDim.x);
  }
}

// ---------------- readout: one wave32 per (b,o) output element ----------------
__global__ void esn_out(const float* __restrict__ Sf, const float* __restrict__ Wout,
                        float* __restrict__ out) {
  const int lane = threadIdx.x & 31;
  const int w    = (blockIdx.x * blockDim.x + threadIdx.x) >> 5;  // global wave id
  if (w < B_ * O_) {
    const int b = w / O_, o = w % O_;
    float acc = 0.0f;
    for (int r = lane; r < R_; r += 32)
      acc += Sf[b * R_ + r] * Wout[o * R_ + r];
#pragma unroll
    for (int s = 16; s > 0; s >>= 1) acc += __shfl_xor(acc, s, 32);
    if (lane == 0) out[w] = acc;
  }
}

extern "C" void kernel_launch(void* const* d_in, const int* in_sizes, int n_in,
                              void* d_out, int out_size, void* d_ws, size_t ws_size,
                              hipStream_t stream) {
  const float* x     = (const float*)d_in[0];  // [B,T,I]
  const float* Win   = (const float*)d_in[1];  // [R,I]
  const float* Wres  = (const float*)d_in[2];  // [R,R]
  const float* Wout  = (const float*)d_in[3];  // [O,R]
  const float* noise = (const float*)d_in[4];  // [B,R]

  float* out    = (float*)d_out;               // [B,O] then [B,T,R]
  float* states = out + B_ * O_;

  char* ws = (char*)d_ws;
  bf16_t*   Wb  = (bf16_t*)(ws + WS_WB);
  bf16_t*   Sb0 = (bf16_t*)(ws + WS_SB0);
  bf16_t*   Sb1 = (bf16_t*)(ws + WS_SB1);
  float*    Sf  = (float*)(ws + WS_SF);
  unsigned* bar = (unsigned*)(ws + WS_BAR);

  const int setup_blocks = (R_ * R_ + 255) / 256;
  esn_setup<<<setup_blocks, 256, 0, stream>>>(Wres, Wb, Sb0, Sb1, bar);
  esn_steps<<<NBLK, NTHR, 0, stream>>>(x, Win, noise, Wb, Sb0, Sb1, Sf, states, bar);
  // 96 output dot-products, one wave each: 12 blocks x 8 waves
  esn_out<<<12, 256, 0, stream>>>(Sf, Wout, out);
}
